// GoogleSelfAttention_52029233824263
// MI455X (gfx1250) — matrix operand order repaired and checked by reading, compiled, not run on hardware
//
#include <hip/hip_runtime.h>
#include <hip/hip_bf16.h>

typedef unsigned short u16;
typedef __attribute__((ext_vector_type(16))) __bf16          v16bf;
typedef __attribute__((ext_vector_type(8)))  float           v8f;
typedef __attribute__((ext_vector_type(4)))  float           v4f;
typedef __attribute__((ext_vector_type(16))) unsigned short  v16u;
typedef __attribute__((ext_vector_type(8)))  unsigned short  v8u;

// ---------------- helpers ----------------

__device__ __forceinline__ u16 f2bf(float f) {
    unsigned u = __builtin_bit_cast(unsigned, f);
    u += 0x7FFFu + ((u >> 16) & 1u);          // round-to-nearest-even
    return (u16)(u >> 16);
}
__device__ __forceinline__ float bf2f(u16 h) {
    unsigned u = ((unsigned)h) << 16;
    return __builtin_bit_cast(float, u);
}
// packed fp32x2 -> bf16x2
__device__ __forceinline__ unsigned f2bf2(float lo, float hi) {
#if __has_builtin(__builtin_amdgcn_cvt_pk_bf16_f32)
    auto p = __builtin_amdgcn_cvt_pk_bf16_f32(lo, hi);
    return __builtin_bit_cast(unsigned, p);
#else
    return (unsigned)f2bf(lo) | ((unsigned)f2bf(hi) << 16);
#endif
}

// async copy: 16 bytes global -> LDS, tracked by ASYNCcnt
__device__ __forceinline__ void async_load_b128(unsigned lds_off, const void* gptr) {
    asm volatile("global_load_async_to_lds_b128 %0, %1, off"
                 :: "v"(lds_off), "v"(gptr) : "memory");
}
__device__ __forceinline__ void wait_asynccnt0() {
    asm volatile("s_wait_asynccnt 0x0" ::: "memory");
}

// A-fragment (16x32 bf16, MxK) from bf16 row-major source (global or LDS).
// Contiguous-K order used here (matches load_B_T): lanes 0-15 = rows, hold K kb+0..15;
//             lanes 16-31 hold K kb+16..31.
__device__ __forceinline__ v16bf load_A_bf16(const u16* base, int ld,
                                             int rowbase, int kb, int lane) {
    int half = (lane >> 4) & 1, r = lane & 15;
    const u16* p = base + (size_t)(rowbase + r) * ld + kb + half * 16;
    struct { v8u lo, hi; } t;
    t.lo = *(const v8u*)(p);
    t.hi = *(const v8u*)(p + 8);
    return __builtin_bit_cast(v16bf, t);
}

// B-fragment (32x16 bf16, KxN) from a K-contiguous ("transposed") source:
// baseT is [Ncols][ld] row-major, row = output column. Contiguous-K order (matches load_A_bf16):
// lanes 0-15 = col n, K kb+0..15; lanes 16-31 = col n, K kb+16..31.
__device__ __forceinline__ v16bf load_B_T(const u16* baseT, int ld,
                                          int colbase, int kb, int lane) {
    int half = (lane >> 4) & 1, c = lane & 15;
    const u16* p = baseT + (size_t)(colbase + c) * ld + kb + half * 16;
    v16u v = *(const v16u*)(p);
    return __builtin_bit_cast(v16bf, v);
}

__device__ __forceinline__ v8f wmma_bf16(v16bf a, v16bf b, v8f c) {
    return __builtin_amdgcn_wmma_f32_16x16x32_bf16(
        false, a, false, b, (short)0, c, false, false);
}

// ---------------- kernels ----------------

// x fp32 -> bf16, one-shot (8 elements / thread)
__global__ void k_cvt_x(const float* __restrict__ X, u16* __restrict__ Xb) {
    int idx = blockIdx.x * blockDim.x + threadIdx.x;
    const float* p = X + (size_t)idx * 8;
    v4f a = *(const v4f*)p;
    v4f b = *(const v4f*)(p + 4);
    uint4 o;
    o.x = f2bf2(a[0], a[1]); o.y = f2bf2(a[2], a[3]);
    o.z = f2bf2(b[0], b[1]); o.w = f2bf2(b[2], b[3]);
    *(uint4*)(Xb + (size_t)idx * 8) = o;
}

// W [K, Nout] fp32 -> WT [Nout, K] bf16
__global__ void k_transpose_w(const float* __restrict__ W, u16* __restrict__ WT,
                              int K, int Nout) {
    int idx = blockIdx.x * blockDim.x + threadIdx.x;
    if (idx >= K * Nout) return;
    int n = idx / K, k = idx - n * K;
    WT[idx] = f2bf(W[(size_t)k * Nout + n]);
}

// Y[r, n] = sum_k Xb[r,k] * WT[n,k];  Xb bf16 [R,K], WT bf16 [Nout,K], Y bf16.
// Block = 256 thr = 8 waves; wave tile 16 rows x 64 cols.
__global__ void k_gemm_proj(const u16* __restrict__ Xb, const u16* __restrict__ WT,
                            u16* __restrict__ Y, int K, int Nout) {
    int lane = threadIdx.x & 31, wave = threadIdx.x >> 5;
    int rowbase = (blockIdx.x * 8 + wave) * 16;
    int colbase = blockIdx.y * 64;
    v8f acc[4] = {};
    for (int kb = 0; kb < K; kb += 32) {
        v16bf a = load_A_bf16(Xb, K, rowbase, kb, lane);
#pragma unroll
        for (int t = 0; t < 4; ++t) {
            v16bf b = load_B_T(WT, K, colbase + t * 16, kb, lane);
            acc[t] = wmma_bf16(a, b, acc[t]);
        }
    }
    int half = lane >> 4, nc = lane & 15;
#pragma unroll
    for (int t = 0; t < 4; ++t)
#pragma unroll
        for (int r = 0; r < 8; ++r)
            Y[(size_t)(rowbase + half * 8 + r) * Nout + colbase + t * 16 + nc] =
                f2bf(acc[t][r]);
}

// 2x2 maxpool on f_conv [8,64,64,64] -> Fp [8,32,32,64]
__global__ void k_pool_f(const u16* __restrict__ FC, u16* __restrict__ FP) {
    int idx = blockIdx.x * blockDim.x + threadIdx.x;
    if (idx >= 8 * 32 * 32 * 64) return;
    int c = idx & 63, xo = (idx >> 6) & 31, yo = (idx >> 11) & 31, b = idx >> 16;
    const u16* p = FC + ((size_t)b * 262144) + (size_t)(yo * 2) * 4096 + (xo * 2) * 64 + c;
    float m = bf2f(p[0]);
    m = fmaxf(m, bf2f(p[64]));
    m = fmaxf(m, bf2f(p[4096]));
    m = fmaxf(m, bf2f(p[4096 + 64]));
    FP[idx] = f2bf(m);
}

// 2x2 maxpool + transpose: h_conv [8,64,64,256] -> HT [8,256,1024]
__global__ void k_pool_h(const u16* __restrict__ HC, u16* __restrict__ HT) {
    int idx = blockIdx.x * blockDim.x + threadIdx.x;
    if (idx >= 8 * 32 * 32 * 256) return;
    int c = idx & 255, xo = (idx >> 8) & 31, yo = (idx >> 13) & 31, b = idx >> 18;
    const u16* p = HC + ((size_t)b * 1048576) + (size_t)(yo * 2) * 16384 + (xo * 2) * 256 + c;
    float m = bf2f(p[0]);
    m = fmaxf(m, bf2f(p[256]));
    m = fmaxf(m, bf2f(p[16384]));
    m = fmaxf(m, bf2f(p[16384 + 256]));
    HT[((size_t)b * 256 + c) * 1024 + (yo * 32 + xo)] = f2bf(m);
}

// Flash attention: softmax(g f^T) h, per batch.
// Block = 4 waves = 64 query rows; each wave owns 16 query rows and the FULL
// Dv=256 accumulator (16 C-tiles). K (32x64) and V (256x32) tiles are shared
// by all waves, staged double-buffered via GLOBAL_LOAD_ASYNC_TO_LDS_B128.
__global__ void __launch_bounds__(128)
k_attn(const u16* __restrict__ G, const u16* __restrict__ F,
       const u16* __restrict__ HT, u16* __restrict__ O) {
    __shared__ __align__(64) u16 Ftile[2][32 * 64];    // 2 x 4 KB
    __shared__ __align__(64) u16 Vtile[2][256 * 32];   // 2 x 16 KB
    __shared__ __align__(64) u16 Pbuf[4][16 * 32];     // 4 KB
    int lane = threadIdx.x & 31, wave = threadIdx.x >> 5, tid = threadIdx.x;
    int b = blockIdx.y;
    int qb = (blockIdx.x * 4 + wave) * 16;
    const u16* gb = G + (size_t)b * 4096 * 64;
    const u16* fb = F + (size_t)b * 1024 * 64;
    const u16* hb = HT + (size_t)b * 256 * 1024;
    u16* ob = O + (size_t)b * 4096 * 256;
    int half = lane >> 4, lc = lane & 15;

    v16bf q0 = load_A_bf16(gb, 64, qb, 0, lane);
    v16bf q1 = load_A_bf16(gb, 64, qb, 32, lane);

    float rmax[8], rsum[8];
#pragma unroll
    for (int r = 0; r < 8; ++r) { rmax[r] = -3.3895314e38f; rsum[r] = 0.f; }
    v8f acc[16] = {};
    u16* P = &Pbuf[wave][0];

    auto stage = [&](int mb, int buf) {
        unsigned fbase = (unsigned)(uintptr_t)&Ftile[buf][0];
        unsigned vbase = (unsigned)(uintptr_t)&Vtile[buf][0];
        // K tile f[mb..mb+32) x 64ch: 4 KB = 256 x b128, 128 thr x 2
#pragma unroll
        for (int i = 0; i < 2; ++i) {
            int c = tid + i * 128;
            async_load_b128(fbase + c * 16, fb + (size_t)mb * 64 + c * 8);
        }
        // V tile hT[0..256)[mb..mb+32): 16 KB = 1024 x b128, 128 thr x 8
#pragma unroll
        for (int i = 0; i < 8; ++i) {
            int c = tid + i * 128;
            int row = c >> 2, piece = c & 3;
            async_load_b128(vbase + c * 16,
                            hb + (size_t)row * 1024 + mb + piece * 8);
        }
        if (mb + 32 < 1024)
            __builtin_prefetch(fb + (size_t)(mb + 32) * 64, 0, 0);
    };

    stage(0, 0);
    for (int it = 0; it < 32; ++it) {
        int cur = it & 1;
        wait_asynccnt0();        // this wave's copies for buf `cur` landed
        __syncthreads();         // all waves' copies landed; prior reads done
        if (it + 1 < 32) stage((it + 1) * 32, cur ^ 1);   // fill other buffer

        const u16* Fl = &Ftile[cur][0];
        const u16* Vl = &Vtile[cur][0];
        // S = Q K^T for two 16-key tiles (K dim = 64 channels), keys from LDS
        v8f s0 = {}, s1 = {};
        {
            v16bf k00 = load_B_T(Fl, 64, 0,  0,  lane);
            v16bf k01 = load_B_T(Fl, 64, 0,  32, lane);
            v16bf k10 = load_B_T(Fl, 64, 16, 0,  lane);
            v16bf k11 = load_B_T(Fl, 64, 16, 32, lane);
            s0 = wmma_bf16(q0, k00, s0);
            s0 = wmma_bf16(q1, k01, s0);
            s1 = wmma_bf16(q0, k10, s1);
            s1 = wmma_bf16(q1, k11, s1);
        }
        // online softmax over the 32 key columns; row r lives in VGPR r,
        // columns across the 16 lanes of this half-wave.
        v8f scv;
#pragma unroll
        for (int r = 0; r < 8; ++r) {
            float mx = fmaxf(s0[r], s1[r]);
#pragma unroll
            for (int d = 1; d < 16; d <<= 1) mx = fmaxf(mx, __shfl_xor(mx, d, 32));
            float nm = fmaxf(rmax[r], mx);
            float sc = __expf(rmax[r] - nm);
            float e0 = __expf(s0[r] - nm);
            float e1 = __expf(s1[r] - nm);
            float ps = e0 + e1;
#pragma unroll
            for (int d = 1; d < 16; d <<= 1) ps += __shfl_xor(ps, d, 32);
            rsum[r] = rsum[r] * sc + ps;
            rmax[r] = nm;
            scv[r] = sc;
            int prow = half * 8 + r;
            P[prow * 32 + lc]      = f2bf(e0);   // LDS in-order within wave
            P[prow * 32 + 16 + lc] = f2bf(e1);
        }
#pragma unroll
        for (int t = 0; t < 16; ++t) acc[t] *= scv;
        // P (16x32) as A-fragment; V B-fragments from LDS [256][32]
        v16bf pa = load_A_bf16(P, 32, 0, 0, lane);
#pragma unroll
        for (int t = 0; t < 16; ++t) {
            v16bf vbfr = load_B_T(Vl, 32, t * 16, 0, lane);
            acc[t] = wmma_bf16(pa, vbfr, acc[t]);
        }
    }
    v8f rinv;
#pragma unroll
    for (int r = 0; r < 8; ++r) rinv[r] = 1.0f / rsum[r];
#pragma unroll
    for (int t = 0; t < 16; ++t) {
        v8f vo = acc[t] * rinv;
#pragma unroll
        for (int r = 0; r < 8; ++r)
            ob[(size_t)(qb + half * 8 + r) * 256 + t * 16 + lc] = f2bf(vo[r]);
    }
}

// out = x + gamma * (O · Wo):  O bf16 [32768,256], WoT bf16 [512,256]
__global__ void k_gemm_out(const u16* __restrict__ O, const u16* __restrict__ WoT,
                           const float* __restrict__ X, const float* __restrict__ gamma,
                           float* __restrict__ out) {
    int lane = threadIdx.x & 31, wave = threadIdx.x >> 5;
    int rowbase = (blockIdx.x * 8 + wave) * 16;
    int colbase = blockIdx.y * 64;
    v8f acc[4] = {};
    for (int kb = 0; kb < 256; kb += 32) {
        v16bf a = load_A_bf16(O, 256, rowbase, kb, lane);
#pragma unroll
        for (int t = 0; t < 4; ++t) {
            v16bf b = load_B_T(WoT, 256, colbase + t * 16, kb, lane);
            acc[t] = wmma_bf16(a, b, acc[t]);
        }
    }
    float gm = gamma[0];
    int half = lane >> 4, nc = lane & 15;
#pragma unroll
    for (int t = 0; t < 4; ++t)
#pragma unroll
        for (int r = 0; r < 8; ++r) {
            size_t idx = (size_t)(rowbase + half * 8 + r) * 512 + colbase + t * 16 + nc;
            out[idx] = X[idx] + gm * acc[t][r];
        }
}

// ---------------- launch ----------------

extern "C" void kernel_launch(void* const* d_in, const int* in_sizes, int n_in,
                              void* d_out, int out_size, void* d_ws, size_t ws_size,
                              hipStream_t stream) {
    (void)in_sizes; (void)n_in; (void)out_size; (void)ws_size;
    const float* x     = (const float*)d_in[0];
    const float* wf    = (const float*)d_in[1];
    const float* wg    = (const float*)d_in[2];
    const float* wh    = (const float*)d_in[3];
    const float* wo    = (const float*)d_in[4];
    const float* gamma = (const float*)d_in[5];
    float* out = (float*)d_out;

    char* ws = (char*)d_ws;
    size_t off = 0;
    auto take = [&](size_t bytes) {
        char* p = ws + off;
        off = (off + bytes + 255) & ~(size_t)255;
        return p;
    };
    u16* Xb  = (u16*)take((size_t)32768 * 512 * 2);
    u16* WfT = (u16*)take((size_t)64 * 512 * 2);
    u16* WgT = (u16*)take((size_t)64 * 512 * 2);
    u16* WhT = (u16*)take((size_t)256 * 512 * 2);
    u16* WoT = (u16*)take((size_t)512 * 256 * 2);
    u16* Gm  = (u16*)take((size_t)32768 * 64 * 2);
    u16* Fc  = (u16*)take((size_t)32768 * 64 * 2);
    u16* Hc  = (u16*)take((size_t)32768 * 256 * 2);
    u16* Fp  = (u16*)take((size_t)8 * 1024 * 64 * 2);
    u16* HT  = (u16*)take((size_t)8 * 256 * 1024 * 2);
    u16* Ob  = (u16*)take((size_t)32768 * 256 * 2);

    k_cvt_x<<<8192, 256, 0, stream>>>(x, Xb);
    k_transpose_w<<<(64 * 512 + 255) / 256, 256, 0, stream>>>(wf, WfT, 512, 64);
    k_transpose_w<<<(64 * 512 + 255) / 256, 256, 0, stream>>>(wg, WgT, 512, 64);
    k_transpose_w<<<(256 * 512 + 255) / 256, 256, 0, stream>>>(wh, WhT, 512, 256);
    k_transpose_w<<<(512 * 256 + 255) / 256, 256, 0, stream>>>(wo, WoT, 256, 512);

    dim3 blk(256);
    k_gemm_proj<<<dim3(256, 1), blk, 0, stream>>>(Xb, WgT, Gm, 512, 64);
    k_gemm_proj<<<dim3(256, 1), blk, 0, stream>>>(Xb, WfT, Fc, 512, 64);
    k_gemm_proj<<<dim3(256, 4), blk, 0, stream>>>(Xb, WhT, Hc, 512, 256);

    k_pool_f<<<(8 * 32 * 32 * 64) / 256, 256, 0, stream>>>(Fc, Fp);
    k_pool_h<<<(8 * 32 * 32 * 256) / 256, 256, 0, stream>>>(Hc, HT);

    k_attn<<<dim3(64, 8), dim3(128), 0, stream>>>(Gm, Fp, HT, Ob);

    k_gemm_out<<<dim3(256, 8), blk, 0, stream>>>(Ob, WoT, x, gamma, out);
}
